// EncoderBlock_26044681682958
// MI455X (gfx1250) — matrix operand order repaired and checked by reading, not compiled, not run on hardware
//
#include <hip/hip_runtime.h>
#include <cstdint>
#include <cstddef>

// ---------------------------------------------------------------------------
// MI455X (gfx1250) transformer encoder block, bf16 WMMA pipeline.
//   B=2, T=4096, D_MODEL=512, H=8, HEAD_DIM=64, D_MLP=2048, M = B*T = 8192
// All GEMMs: D = A(bf16,[M,K]) x Bt(bf16,[N,K])^T  via v_wmma_f32_16x16x32_bf16
// Attention: flash-style; softmax row-max via DPP16 (no LDS), row-sum via an
// extra WMMA against a ones matrix (no cross-lane shuffles at all).
// ---------------------------------------------------------------------------

#define DEV __device__ __forceinline__

typedef __attribute__((ext_vector_type(16))) __bf16 v16bf;
typedef __attribute__((ext_vector_type(8)))  __bf16 v8bf;
typedef __attribute__((ext_vector_type(8)))  float  v8f;

DEV v8f vzero8() {
    v8f z;
#pragma unroll
    for (int i = 0; i < 8; ++i) z[i] = 0.0f;
    return z;
}

DEV v16bf combine16(v8bf lo, v8bf hi) {
    v16bf r;
#pragma unroll
    for (int i = 0; i < 8; ++i) { r[i] = lo[i]; r[8 + i] = hi[i]; }
    return r;
}

DEV v8f wmma_bf16(v16bf a, v16bf b, v8f c) {
    // (neg_a, A, neg_b, B, c_mod, C, reuse_a, reuse_b)
    return __builtin_amdgcn_wmma_f32_16x16x32_bf16(false, a, false, b,
                                                   (short)0, c, false, false);
}

// A fragment: 16x32 bf16 tile, row-major source with leading dim lda (elems).
// Contiguous k order used here (not the ISA table order): lane L (m=L&15, h=L>>4) holds K = h*16 + [0..15].
DEV v16bf load_a_frag(const __bf16* __restrict__ tile, int lda) {
    const int l = threadIdx.x & 31;
    const int m = l & 15, h = l >> 4;
    const __bf16* row = tile + (size_t)m * lda;
    v8bf lo = *(const v8bf*)(row + h * 16);
    v8bf hi = *(const v8bf*)(row + h * 16 + 8);
    return combine16(lo, hi);
}

// B fragment: 32x16 bf16 tile, source stored transposed Bt[n][k] with leading
// dim ldb. Contiguous k order used here, as in A: lane L (n=L&15, h=L>>4) holds K = h*16 + [0..15].
DEV v16bf load_b_frag(const __bf16* __restrict__ tileT, int ldb) {
    const int l = threadIdx.x & 31;
    const int n = l & 15, h = l >> 4;
    const __bf16* p = tileT + (size_t)n * ldb + h * 16;
    v8bf lo = *(const v8bf*)(p);
    v8bf hi = *(const v8bf*)(p + 8);
    return combine16(lo, hi);
}

// ---- 16-lane-group max reduction ------------------------------------------
// Preferred: DPP16 (pure VALU, stays within each 16-lane row, no LDS).
//   quad_perm[1,0,3,2] (0xB1) -> pairwise within quads
//   quad_perm[2,3,0,1] (0x4E) -> quads fully reduced
//   row_half_mirror    (0x141)-> 8-lane groups reduced
//   row_mirror         (0x140)-> 16-lane groups reduced
#if __has_builtin(__builtin_amdgcn_update_dpp)
template <int CTRL>
DEV float dppmax_step(float v) {
    union { float f; int i; } u, r;
    u.f = v;
    r.i = __builtin_amdgcn_update_dpp(u.i, u.i, CTRL, 0xf, 0xf, true);
    return fmaxf(v, r.f);
}
DEV float rmax16(float v) {
    v = dppmax_step<0xB1>(v);
    v = dppmax_step<0x4E>(v);
    v = dppmax_step<0x141>(v);
    v = dppmax_step<0x140>(v);
    return v;
}
#else
DEV float rmax16(float v) {
#pragma unroll
    for (int m = 1; m < 16; m <<= 1) v = fmaxf(v, __shfl_xor(v, m, 32));
    return v;
}
#endif

// ---------------------------------------------------------------------------
// Pre-pass: f32 -> bf16 elementwise, and f32 [K,N] -> bf16 transposed [N,K].
// ---------------------------------------------------------------------------
__global__ void f32_to_bf16_kernel(const float* __restrict__ in,
                                   __bf16* __restrict__ out, size_t n) {
    size_t i = (size_t)blockIdx.x * blockDim.x + threadIdx.x;
    if (i < n) out[i] = (__bf16)in[i];
}

__global__ void transpose_to_bf16_kernel(const float* __restrict__ W,
                                         __bf16* __restrict__ Wt,
                                         int K, int N) {
    size_t i = (size_t)blockIdx.x * blockDim.x + threadIdx.x;
    if (i < (size_t)K * N) {
        int k = (int)(i / N), n = (int)(i % N);
        Wt[(size_t)n * K + k] = (__bf16)W[i];
    }
}

// ---------------------------------------------------------------------------
// Generic bf16 WMMA GEMM.  Block = 256 threads = 8 waves (4 M x 2 N);
// wave tile 32x32 (4 accumulators), block tile 128x64, K step 32, unroll 2.
// MODE 0: out[row*N+col]   (row-major)
// MODE 1: out[((b*8+h)*T + t)*64 + d]   (Q/K: [B,H,T,64])
// MODE 2: out[((b*8+h)*64 + d)*T + t]   (V transposed: [B,H,64,T])
// ---------------------------------------------------------------------------
template <int MODE>
__global__ __launch_bounds__(256) void gemm_bf16_kernel(
    const __bf16* __restrict__ A, const __bf16* __restrict__ Bt,
    const float* __restrict__ bias, const float* __restrict__ resid,
    float* __restrict__ outf, __bf16* __restrict__ outb,
    int M, int N, int K, int Tdim, int relu) {
    const int lane = threadIdx.x & 31;
    const int w    = threadIdx.x >> 5;
    const int wm   = w >> 1, wn = w & 1;
    const size_t m0 = (size_t)blockIdx.y * 128 + (size_t)wm * 32;
    const size_t n0 = (size_t)blockIdx.x * 64 + (size_t)wn * 32;

    v8f c[2][2];
#pragma unroll
    for (int i = 0; i < 2; ++i)
#pragma unroll
        for (int j = 0; j < 2; ++j) c[i][j] = vzero8();

#pragma unroll 2
    for (int k0 = 0; k0 < K; k0 += 32) {
        v16bf a0 = load_a_frag(A + m0 * K + k0, K);
        v16bf a1 = load_a_frag(A + (m0 + 16) * K + k0, K);
        v16bf b0 = load_b_frag(Bt + n0 * K + k0, K);
        v16bf b1 = load_b_frag(Bt + (n0 + 16) * K + k0, K);
        c[0][0] = wmma_bf16(a0, b0, c[0][0]);
        c[0][1] = wmma_bf16(a0, b1, c[0][1]);
        c[1][0] = wmma_bf16(a1, b0, c[1][0]);
        c[1][1] = wmma_bf16(a1, b1, c[1][1]);
        if (k0 + 32 < K) {  // pull next K slab toward L0 (global_prefetch)
            __builtin_prefetch(A + (m0 + lane) * K + k0 + 32, 0, 3);
            __builtin_prefetch(Bt + (n0 + lane) * K + k0 + 32, 0, 3);
        }
    }

    // C layout: lane (n=L&15, h=L>>4), VGPR r -> row m0+i*16+r+8h, col n0+j*16+n
    const int nn = lane & 15, h = lane >> 4;
#pragma unroll
    for (int i = 0; i < 2; ++i)
#pragma unroll
        for (int j = 0; j < 2; ++j)
#pragma unroll
            for (int r = 0; r < 8; ++r) {
                size_t row = m0 + (size_t)i * 16 + r + 8 * h;
                size_t col = n0 + (size_t)j * 16 + nn;
                float v = c[i][j][r];
                if (bias)  v += bias[col];
                if (relu)  v = fmaxf(v, 0.0f);
                if (resid) v += resid[row * (size_t)N + col];
                size_t off;
                if (MODE == 0) {
                    off = row * (size_t)N + col;
                } else {
                    size_t bb = row / (size_t)Tdim, t = row % (size_t)Tdim;
                    size_t hh = col >> 6, d = col & 63;
                    off = (MODE == 1)
                              ? ((bb * 8 + hh) * (size_t)Tdim + t) * 64 + d
                              : ((bb * 8 + hh) * 64 + d) * (size_t)Tdim + t;
                }
                if (outf) outf[off] = v;
                if (outb) outb[off] = (__bf16)v;
            }
}

// ---------------------------------------------------------------------------
// Flash attention, causal.  Block = 128 threads = 4 independent waves; each
// wave owns 16 query rows.  Key tiles of 32; online softmax; P transposed
// C-layout -> A-fragment through a wave-private LDS tile.  Row-sums come from
// an extra WMMA against a ones matrix (result replicated across all columns).
//   q,k : [B,H,T,64] bf16 (row-major per head)     (B-frags of K come free)
//   vt  : [B,H,64,T] bf16 (transposed)             (B-frags of V come free)
//   ctx : [B*T, 512] bf16 row-major (heads merged)
// ---------------------------------------------------------------------------
__global__ __launch_bounds__(128) void flash_attn_kernel(
    const __bf16* __restrict__ q, const __bf16* __restrict__ k,
    const __bf16* __restrict__ vt, __bf16* __restrict__ ctx, int T) {
    __shared__ __align__(16) __bf16 plds[4][16][32];

    const int lane = threadIdx.x & 31;
    const int w    = threadIdx.x >> 5;
    const int bh   = blockIdx.y;
    const int b    = bh >> 3;
    const int hd   = bh & 7;
    const int q0   = blockIdx.x * 64 + w * 16;

    const __bf16* qb = q  + (size_t)bh * T * 64;
    const __bf16* kb = k  + (size_t)bh * T * 64;
    const __bf16* vb = vt + (size_t)bh * T * 64;

    // Q fragments for this wave's 16 rows (d-chunks 0..31, 32..63)
    v16bf aq0 = load_a_frag(qb + (size_t)q0 * 64, 64);
    v16bf aq1 = load_a_frag(qb + (size_t)q0 * 64 + 32, 64);

    // all-ones B matrix for row-sum WMMA
    v16bf bones;
#pragma unroll
    for (int i = 0; i < 16; ++i) bones[i] = (__bf16)1.0f;

    v8f o[4];
#pragma unroll
    for (int dc = 0; dc < 4; ++dc) o[dc] = vzero8();
    float mrow[8], lrow[8];
#pragma unroll
    for (int r = 0; r < 8; ++r) { mrow[r] = -__builtin_inff(); lrow[r] = 0.0f; }

    const int   nn    = lane & 15, h = lane >> 4;
    const float scale = 0.125f;  // 1/sqrt(64)
    const int   jmax  = q0 + 15; // last (largest) query row of this wave

    for (int j = 0; j <= jmax; j += 32) {
        // ---- S = Q x K^T for 32 keys: two 16x16 C tiles, 4 WMMAs
        v8f s[2];
#pragma unroll
        for (int jj = 0; jj < 2; ++jj) {
            const __bf16* kt = kb + (size_t)(j + jj * 16) * 64;
            v16bf bk0 = load_b_frag(kt, 64);       // d = 0..31
            v16bf bk1 = load_b_frag(kt + 32, 64);  // d = 32..63
            v8f t = vzero8();
            t = wmma_bf16(aq0, bk0, t);
            t = wmma_bf16(aq1, bk1, t);
            s[jj] = t;
        }
        if (j + 32 <= jmax) {  // warm next K/V tiles (global_prefetch)
            __builtin_prefetch(kb + (size_t)(j + 32 + lane) * 64, 0, 3);
            __builtin_prefetch(vb + (size_t)lane * T + j + 32, 0, 3);
        }
        // ---- scale + causal mask
#pragma unroll
        for (int jj = 0; jj < 2; ++jj)
#pragma unroll
            for (int r = 0; r < 8; ++r) {
                int qrow = q0 + r + 8 * h;
                int kcol = j + jj * 16 + nn;
                float v = s[jj][r] * scale;
                s[jj][r] = (kcol <= qrow) ? v : -__builtin_inff();
            }
        // ---- online softmax (row max via DPP16; no LDS traffic)
        float alpha[8];
#pragma unroll
        for (int r = 0; r < 8; ++r) {
            float mx = rmax16(fmaxf(s[0][r], s[1][r]));
            float nm = fmaxf(mrow[r], mx);
            alpha[r] = __expf(mrow[r] - nm);  // first tile: exp(-inf)=0
            mrow[r]  = nm;
        }
#pragma unroll
        for (int jj = 0; jj < 2; ++jj)
#pragma unroll
            for (int r = 0; r < 8; ++r)
                s[jj][r] = __expf(s[jj][r] - mrow[r]);
#pragma unroll
        for (int dc = 0; dc < 4; ++dc)
#pragma unroll
            for (int r = 0; r < 8; ++r) o[dc][r] *= alpha[r];

        // ---- P (C layout) -> LDS (row-major 16x32) -> A fragment
#pragma unroll
        for (int jj = 0; jj < 2; ++jj)
#pragma unroll
            for (int r = 0; r < 8; ++r)
                plds[w][r + 8 * h][jj * 16 + nn] = (__bf16)s[jj][r];
        asm volatile("s_wait_dscnt 0" ::: "memory");
        {
            const int m = lane & 15, hh = lane >> 4;
            v8bf lo = *(const v8bf*)&plds[w][m][hh * 16];
            v8bf hi = *(const v8bf*)&plds[w][m][hh * 16 + 8];
            v16bf pa = combine16(lo, hi);

            // row-sum of P via WMMA against ones: every column = rowsum
            v8f rs = wmma_bf16(pa, bones, vzero8());
#pragma unroll
            for (int r = 0; r < 8; ++r)
                lrow[r] = lrow[r] * alpha[r] + rs[r];

            // ---- O += P x V (4 d-chunks of 16), B-frags from vt[d][t]
#pragma unroll
            for (int dc = 0; dc < 4; ++dc) {
                const __bf16* vp =
                    vb + (size_t)(dc * 16 + nn) * T + j + h * 16;
                v16bf bv = combine16(*(const v8bf*)vp, *(const v8bf*)(vp + 8));
                o[dc] = wmma_bf16(pa, bv, o[dc]);
            }
        }
    }

    // ---- normalize + store ctx as bf16 [B*T, 512] with heads merged
#pragma unroll
    for (int r = 0; r < 8; ++r) lrow[r] = 1.0f / lrow[r];
#pragma unroll
    for (int dc = 0; dc < 4; ++dc)
#pragma unroll
        for (int r = 0; r < 8; ++r) {
            size_t t   = (size_t)q0 + r + 8 * h;
            size_t off = ((size_t)b * T + t) * 512 + (size_t)hd * 64 +
                         dc * 16 + nn;
            ctx[off] = (__bf16)(o[dc][r] * lrow[r]);
        }
}

// ---------------------------------------------------------------------------
// Launch
// ---------------------------------------------------------------------------
extern "C" void kernel_launch(void* const* d_in, const int* in_sizes, int n_in,
                              void* d_out, int out_size, void* d_ws,
                              size_t ws_size, hipStream_t stream) {
    (void)in_sizes; (void)n_in; (void)out_size; (void)ws_size;
    constexpr int B = 2, T = 4096, D = 512, H = 8, DMLP = 2048;
    constexpr size_t M = (size_t)B * T;  // 8192

    const float* x  = (const float*)d_in[0];
    const float* Wq = (const float*)d_in[1];
    const float* bq = (const float*)d_in[2];
    const float* Wk = (const float*)d_in[3];
    const float* bk = (const float*)d_in[4];
    const float* Wv = (const float*)d_in[5];
    const float* bv = (const float*)d_in[6];
    const float* Wo = (const float*)d_in[7];
    const float* bo = (const float*)d_in[8];
    const float* W1 = (const float*)d_in[9];
    const float* b1 = (const float*)d_in[10];
    const float* W2 = (const float*)d_in[11];
    const float* b2 = (const float*)d_in[12];
    float* out = (float*)d_out;

    // workspace carve-up (256B aligned)
    uint8_t* p = (uint8_t*)d_ws;
    auto alloc = [&](size_t bytes) -> void* {
        void* r = (void*)p;
        p += (bytes + 255) & ~(size_t)255;
        return r;
    };
    __bf16* xb   = (__bf16*)alloc(M * D * 2);
    __bf16* Wqt  = (__bf16*)alloc((size_t)D * D * 2);
    __bf16* Wkt  = (__bf16*)alloc((size_t)D * D * 2);
    __bf16* Wvt  = (__bf16*)alloc((size_t)D * D * 2);
    __bf16* Wot  = (__bf16*)alloc((size_t)D * D * 2);
    __bf16* W1t  = (__bf16*)alloc((size_t)D * DMLP * 2);
    __bf16* W2t  = (__bf16*)alloc((size_t)DMLP * D * 2);
    __bf16* qb   = (__bf16*)alloc(M * D * 2);   // [B,H,T,64]
    __bf16* kbuf = (__bf16*)alloc(M * D * 2);   // [B,H,T,64]
    __bf16* vtb  = (__bf16*)alloc(M * D * 2);   // [B,H,64,T]
    __bf16* ctxb = (__bf16*)alloc(M * D * 2);   // [B*T,512]
    float*  x2f  = (float*)alloc(M * D * 4);
    __bf16* x2b  = (__bf16*)alloc(M * D * 2);
    __bf16* hb   = (__bf16*)alloc(M * DMLP * 2);

    // 1) convert x, transpose weights to bf16 [N,K]
    {
        size_t n = M * D;
        f32_to_bf16_kernel<<<(unsigned)((n + 255) / 256), 256, 0, stream>>>(x, xb, n);
        size_t nw = (size_t)D * D;
        transpose_to_bf16_kernel<<<(unsigned)((nw + 255) / 256), 256, 0, stream>>>(Wq, Wqt, D, D);
        transpose_to_bf16_kernel<<<(unsigned)((nw + 255) / 256), 256, 0, stream>>>(Wk, Wkt, D, D);
        transpose_to_bf16_kernel<<<(unsigned)((nw + 255) / 256), 256, 0, stream>>>(Wv, Wvt, D, D);
        transpose_to_bf16_kernel<<<(unsigned)((nw + 255) / 256), 256, 0, stream>>>(Wo, Wot, D, D);
        size_t n1 = (size_t)D * DMLP;
        transpose_to_bf16_kernel<<<(unsigned)((n1 + 255) / 256), 256, 0, stream>>>(W1, W1t, D, DMLP);
        transpose_to_bf16_kernel<<<(unsigned)((n1 + 255) / 256), 256, 0, stream>>>(W2, W2t, DMLP, D);
    }

    // 2) QKV projections (bias, bf16 out, head-split layouts)
    dim3 blk(256);
    dim3 g512(512 / 64, (unsigned)(M / 128));
    gemm_bf16_kernel<1><<<g512, blk, 0, stream>>>(xb, Wqt, bq, nullptr,
                                                  nullptr, qb, (int)M, D, D, T, 0);
    gemm_bf16_kernel<1><<<g512, blk, 0, stream>>>(xb, Wkt, bk, nullptr,
                                                  nullptr, kbuf, (int)M, D, D, T, 0);
    gemm_bf16_kernel<2><<<g512, blk, 0, stream>>>(xb, Wvt, bv, nullptr,
                                                  nullptr, vtb, (int)M, D, D, T, 0);

    // 3) causal flash attention -> ctx bf16 [B*T,512]
    flash_attn_kernel<<<dim3(T / 64, B * H), dim3(128), 0, stream>>>(
        qb, kbuf, vtb, ctxb, T);

    // 4) x2 = x + ctx @ Wo + bo   (f32 for later residual + bf16 for next GEMM)
    gemm_bf16_kernel<0><<<g512, blk, 0, stream>>>(ctxb, Wot, bo, x,
                                                  x2f, x2b, (int)M, D, D, T, 0);

    // 5) h = relu(x2 @ W1 + b1)  (bf16)
    dim3 g2048(DMLP / 64, (unsigned)(M / 128));
    gemm_bf16_kernel<0><<<g2048, blk, 0, stream>>>(x2b, W1t, b1, nullptr,
                                                   nullptr, hb, (int)M, DMLP, D, T, 1);

    // 6) out = x2 + h @ W2 + b2  (f32 -> d_out)
    gemm_bf16_kernel<0><<<g512, blk, 0, stream>>>(hb, W2t, b2, x2f,
                                                  out, nullptr, (int)M, D, DMLP, T, 0);
}
